// MechanismCodebook_39565238731283
// MI455X (gfx1250) — compile-verified, matrix operand order checked by reading, not run on hardware
//
#include <hip/hip_runtime.h>
#include <hip/hip_bf16.h>
#include <math.h>

// ---------------------------------------------------------------------------
// MechanismCodebook fused edge-MLP + codebook attention for gfx1250 (MI455X)
// B=256, K=32, D=128, H=256, N=16.  bf16 WMMA (f32 accum), wave32,
// TDM tensor_load_to_lds for all LDS staging in the fused kernel.
// ---------------------------------------------------------------------------

typedef __attribute__((ext_vector_type(16))) __bf16 v16bf;
typedef __attribute__((ext_vector_type(8)))  __bf16 v8bf;
typedef __attribute__((ext_vector_type(8)))  float  v8f;
typedef __attribute__((ext_vector_type(4)))  unsigned int u32x4;
typedef __attribute__((ext_vector_type(8)))  unsigned int u32x8;

union AFrag { v16bf v; v8bf h[2]; __bf16 a[16]; };
union CFrag { v8f  v; float  a[8];  };

// ISA 7.12.2 16-bit fragment: element e<8 -> K = e + 8*hh ; e>=8 -> K = e+8 + 8*hh.
// So e=0..7 is a contiguous 8-element K-run, e=8..15 the run 16 higher.
__device__ __forceinline__ int kOfElem(int e, int hh) {
  return (e < 8 ? e : e + 8) + (hh ? 8 : 0);
}

__device__ __forceinline__ float gelu_exact(float x) {
  return 0.5f * x * (1.0f + erff(x * 0.70710678118654752f));
}

// ---------------------------------------------------------------------------
// TDM: 1-D contiguous tile, global -> LDS.  D# group0/group1 per ISA 8.3/8.4.
// log2sz: 1 = 2-byte elems (bf16), 2 = 4-byte elems (f32).
// ---------------------------------------------------------------------------
__device__ __forceinline__ void tdm_load_1d(void* ldsDst, const void* src,
                                            unsigned nElems, unsigned log2sz) {
  const unsigned long long ga = (unsigned long long)(uintptr_t)src;
  const unsigned lds = (unsigned)(uintptr_t)ldsDst;   // low 32 bits = LDS offset
  u32x4 g0;
  g0[0] = 1u;                                   // count=1 (valid user D#)
  g0[1] = lds;                                  // lds_addr
  g0[2] = (unsigned)ga;                         // global_addr[31:0]
  g0[3] = (unsigned)(ga >> 32) | (2u << 30);    // global_addr[56:32] | type=2
  u32x8 g1;
  g1[0] = log2sz << 16;                         // data_size (0:1B,1:2B,2:4B)
  g1[1] = (nElems & 0xFFFFu) << 16;             // tensor_dim0[15:0]
  g1[2] = (nElems >> 16) | (1u << 16);          // tensor_dim0[31:16], tensor_dim1=1
  g1[3] = (nElems & 0xFFFFu) << 16;             // tile_dim0 = nElems (<=32768)
  g1[4] = 0u;                                   // tile_dim1=0 (1-D), tile_dim2=0
  g1[5] = nElems;                               // tensor_dim0_stride[31:0]
  g1[6] = 0u;
  g1[7] = 0u;
  asm volatile("tensor_load_to_lds %0, %1" :: "s"(g0), "s"(g1) : "memory");
}

// ---------------------------------------------------------------------------
// Kernel 0: one-time pack.  z -> bf16 rows; W1/W2/codebook -> fragment-order
// bf16 so every later B-fragment is a single contiguous 32-byte load.
// 262144 threads (grid 1024 x 256).
// ---------------------------------------------------------------------------
__global__ void __launch_bounds__(256)
pack_kernel(const float* __restrict__ z, const float* __restrict__ W1,
            const float* __restrict__ W2, const float* __restrict__ cb,
            __bf16* __restrict__ zb, __bf16* __restrict__ w1sw,
            __bf16* __restrict__ w2sw, __bf16* __restrict__ cbm,
            __bf16* __restrict__ cbrow) {
  const int t = blockIdx.x * 256 + threadIdx.x;
  {                                             // z (1048576 f32) -> bf16
    const float4 q = ((const float4*)z)[t];
    union { __bf16 h[4]; unsigned long long u; } p;
    p.h[0] = (__bf16)q.x; p.h[1] = (__bf16)q.y;
    p.h[2] = (__bf16)q.z; p.h[3] = (__bf16)q.w;
    *reinterpret_cast<unsigned long long*>(zb + 4 * (size_t)t) = p.u;
  }
  if (t < 65536) {                              // W1 -> frag order [nt32][kk4][lane][e]
    const int e = t & 15, lane = (t >> 4) & 31, kk = (t >> 9) & 3, nt = t >> 11;
    const int hh = lane >> 4, ln = lane & 15;
    const int k = kk * 32 + kOfElem(e, hh);
    const int col = nt * 16 + ln;               // 0..255 h_i cols, 256..511 h_j
    const int csel = col >> 8;                  // branchless hi/hj split
    w1sw[t] = (__bf16)W1[(size_t)(k + (csel << 7)) * 256 + (col & 255)];
  }
  if (t < 32768) {                              // W2 -> frag order [kk8][nt8][lane][e]
    const int e = t & 15, lane = (t >> 4) & 31, nt = (t >> 9) & 7, kk = t >> 12;
    const int hh = lane >> 4, ln = lane & 15;
    const int k = kk * 32 + kOfElem(e, hh);
    w2sw[t] = (__bf16)W2[k * 128 + nt * 16 + ln];
  }
  if (t < 4096) {                               // cb as m-GEMM B frags, K padded to 32
    const int e = t & 15, lane = (t >> 4) & 31, nt = t >> 9;
    const int hh = lane >> 4, ln = lane & 15;
    const int k = kOfElem(e, hh);
    cbm[t] = (k < 16) ? (__bf16)cb[k * 128 + nt * 16 + ln] : (__bf16)0.0f;
  }
  if (t < 2048) cbrow[t] = (__bf16)cb[t];       // cb row-major bf16 (logits B)
}

// ---------------------------------------------------------------------------
// Kernel 1: [h_i | h_j] = z @ [W1[:D] | W1[D:]]  (8192x128 @ 128x512), fp32 out.
// LDS-free: A frags = 16B global loads of zb rows, B frags = 32B loads of w1sw.
// ---------------------------------------------------------------------------
__global__ void __launch_bounds__(256)
hij_gemm_kernel(const __bf16* __restrict__ zb, const __bf16* __restrict__ w1sw,
                float* __restrict__ hi, float* __restrict__ hj) {
  const int tid = threadIdx.x;
  const int g0  = blockIdx.x * 16;
  const int w = tid >> 5, lane = tid & 31, hh = lane >> 4, ln = lane & 15;
  const __bf16* zrow = zb + (size_t)(g0 + ln) * 128;
#pragma unroll 1
  for (int ntj = 0; ntj < 4; ++ntj) {
    const int nt = w * 4 + ntj;
    const int col = nt * 16 + ln;
    const int csel = col >> 8;
    CFrag acc;
#pragma unroll
    for (int v = 0; v < 8; ++v) acc.a[v] = 0.0f;
#pragma unroll
    for (int kk = 0; kk < 4; ++kk) {
      AFrag A, Bf;
      A.h[0] = *(const v8bf*)(zrow + kk * 32 + 8 * hh);
      A.h[1] = *(const v8bf*)(zrow + kk * 32 + 16 + 8 * hh);
      Bf.v   = *(const v16bf*)(w1sw + (size_t)((nt * 4 + kk) * 32 + lane) * 16);
      acc.v = __builtin_amdgcn_wmma_f32_16x16x32_bf16(
          false, A.v, false, Bf.v, (short)0, acc.v, false, false);
    }
    float* dst = csel ? hj : hi;                // branchless pointer select
    const int cc = col & 255;
#pragma unroll
    for (int v = 0; v < 8; ++v) {
      const int m = v + (hh ? 8 : 0);
      dst[(size_t)(g0 + m) * 256 + cc] = acc.a[v];
    }
  }
}

// ---------------------------------------------------------------------------
// Kernel 2: per (b, i-tile, j-tile): e = gelu(h)@W2 ; logits = e@cb^T/T ;
// softmax ; m = alpha@cb.   Entire LDS staging via 7 TDM descriptors.
// ---------------------------------------------------------------------------
__global__ void __launch_bounds__(256)
edge_fused_kernel(const float* __restrict__ hiW, const float* __restrict__ hjW,
                  const float* __restrict__ b1, const float* __restrict__ b2,
                  const __bf16* __restrict__ w2sw, const __bf16* __restrict__ cbm,
                  const __bf16* __restrict__ cbrow,
                  float* __restrict__ mOut, float* __restrict__ aOut,
                  float* __restrict__ eOut, float* __restrict__ lOut) {
  extern __shared__ __align__(16) char smem[];
  float*  hiS  = (float*)smem;           // 16x256 f32
  float*  hjS  = hiS + 4096;             // 16x256 f32
  float*  b1S  = hjS + 4096;             // 256
  float*  b2S  = b1S + 256;              // 128
  __bf16* w2S  = (__bf16*)(b2S + 128);   // 32768 frag-order
  __bf16* cbmS = w2S + 32768;            // 4096 frag-order (zero-padded K)
  __bf16* cbrS = cbmS + 4096;            // 2048 row-major
  __bf16* stg  = cbrS + 2048;            // 8 waves x 2048 staging

  const int tid = threadIdx.x;
  const int b = blockIdx.x, it = blockIdx.y, jt = blockIdx.z;

  if (tid < 32) {                        // wave 0 drives the Tensor Data Mover
    tdm_load_1d(hiS, hiW + ((size_t)b * 32 + it * 16) * 256, 4096, 2);
    tdm_load_1d(hjS, hjW + ((size_t)b * 32 + jt * 16) * 256, 4096, 2);
    tdm_load_1d(b1S, b1, 256, 2);
    tdm_load_1d(b2S, b2, 128, 2);
    tdm_load_1d(w2S, w2sw, 32768, 1);
    tdm_load_1d(cbmS, cbm, 4096, 1);
    tdm_load_1d(cbrS, cbrow, 2048, 1);
    __builtin_amdgcn_s_wait_tensorcnt(0);
  }
  __syncthreads();

  const int w = tid >> 5, lane = tid & 31, hh = lane >> 4, ln = lane & 15;
  __bf16* stgW = stg + w * 2048;

#pragma unroll 1
  for (int ri = 0; ri < 2; ++ri) {
    const int rt = w * 2 + ri;           // i_local (rows of tile share i)
    const int i  = it * 16 + rt;

    // ---- e = gelu(h_i[rt] + h_j[m] + b1) @ W2  (16x256 @ 256x128) ----
    CFrag accE[8];
#pragma unroll
    for (int nt = 0; nt < 8; ++nt)
#pragma unroll
      for (int v = 0; v < 8; ++v) accE[nt].a[v] = 0.0f;
#pragma unroll 1
    for (int kk = 0; kk < 8; ++kk) {
      AFrag A;
#pragma unroll
      for (int e = 0; e < 16; ++e) {
        const int k = kk * 32 + kOfElem(e, hh);
        const float hv = hiS[rt * 256 + k] + hjS[ln * 256 + k] + b1S[k];
        A.a[e] = (__bf16)gelu_exact(hv);
      }
#pragma unroll
      for (int nt = 0; nt < 8; ++nt) {
        AFrag Bf;
        Bf.v = *(const v16bf*)(w2S + ((kk * 8 + nt) * 32 + lane) * 16);
        accE[nt].v = __builtin_amdgcn_wmma_f32_16x16x32_bf16(
            false, A.v, false, Bf.v, (short)0, accE[nt].v, false, false);
      }
    }

    // finalize e: +b2, store fp32, stage bf16 rows for the logits GEMM
#pragma unroll
    for (int nt = 0; nt < 8; ++nt) {
      const int n = nt * 16 + ln;
      const float bb = b2S[n];
#pragma unroll
      for (int v = 0; v < 8; ++v) {
        const int m = v + (hh ? 8 : 0);
        const int j = jt * 16 + m;
        const float val = accE[nt].a[v] + bb;
        eOut[(((size_t)b * 32 + i) * 32 + j) * 128 + n] = val;
        stgW[m * 128 + n] = (__bf16)val;
      }
    }
    __syncthreads();                     // C-layout -> A-layout visibility

    // ---- logits = (e @ cb^T) / T   (16x128 @ 128x16) ----
    CFrag accL;
#pragma unroll
    for (int v = 0; v < 8; ++v) accL.a[v] = 0.0f;
#pragma unroll
    for (int kk = 0; kk < 4; ++kk) {
      AFrag A, Bf;
      const __bf16* eRow = stgW + ln * 128 + kk * 32 + 8 * hh;
      const __bf16* cRow = cbrS + ln * 128 + kk * 32 + 8 * hh;
      A.h[0]  = *(const v8bf*)(eRow);
      A.h[1]  = *(const v8bf*)(eRow + 16);
      Bf.h[0] = *(const v8bf*)(cRow);
      Bf.h[1] = *(const v8bf*)(cRow + 16);
      accL.v = __builtin_amdgcn_wmma_f32_16x16x32_bf16(
          false, A.v, false, Bf.v, (short)0, accL.v, false, false);
    }

    // ---- softmax over 16 codes (codes sit across each 16-lane half) ----
    const int c = ln;
#pragma unroll
    for (int v = 0; v < 8; ++v) {
      const int m = v + (hh ? 8 : 0);
      const int j = jt * 16 + m;
      const float lg = accL.a[v] * 10.0f;          // 1 / TEMPERATURE
      lOut[(((size_t)b * 32 + i) * 32 + j) * 16 + c] = lg;
      float mx = lg;
#pragma unroll
      for (int msk = 1; msk < 16; msk <<= 1) mx = fmaxf(mx, __shfl_xor(mx, msk, 16));
      const float ex = __expf(lg - mx);
      float sm = ex;
#pragma unroll
      for (int msk = 1; msk < 16; msk <<= 1) sm += __shfl_xor(sm, msk, 16);
      const float al = ex / sm;
      aOut[(((size_t)b * 32 + i) * 32 + j) * 16 + c] = al;
      stgW[m * 16 + c] = (__bf16)al;               // stage alpha (reuse stg)
    }
    __syncthreads();

    // ---- m = alpha @ cb   (16x16 @ 16x128, K zero-padded to 32) ----
    const v8bf zz = {};
#pragma unroll
    for (int nt = 0; nt < 8; ++nt) {
      AFrag A, Bf;
      A.h[0] = *(const v8bf*)(stgW + ln * 16 + 8 * hh);
      A.h[1] = zz;                                 // K=16..31 are zero
      Bf.v = *(const v16bf*)(cbmS + (nt * 32 + lane) * 16);
      CFrag accM;
#pragma unroll
      for (int v = 0; v < 8; ++v) accM.a[v] = 0.0f;
      accM.v = __builtin_amdgcn_wmma_f32_16x16x32_bf16(
          false, A.v, false, Bf.v, (short)0, accM.v, false, false);
#pragma unroll
      for (int v = 0; v < 8; ++v) {
        const int m = v + (hh ? 8 : 0);
        const int j = jt * 16 + m;
        mOut[(((size_t)b * 32 + i) * 32 + j) * 128 + nt * 16 + ln] = accM.a[v];
      }
    }
    __syncthreads();                     // staging reuse next iteration
  }
}

// ---------------------------------------------------------------------------
extern "C" void kernel_launch(void* const* d_in, const int* in_sizes, int n_in,
                              void* d_out, int out_size, void* d_ws, size_t ws_size,
                              hipStream_t stream) {
  const float* z  = (const float*)d_in[0];
  const float* W1 = (const float*)d_in[1];
  const float* b1 = (const float*)d_in[2];
  const float* W2 = (const float*)d_in[3];
  const float* b2 = (const float*)d_in[4];
  const float* cb = (const float*)d_in[5];

  float* out  = (float*)d_out;
  float* mOut = out;                      // (B,K,K,D)
  float* aOut = out + 33554432ull;        // (B,K,K,N)
  float* eOut = out + 37748736ull;        // (B,K,K,D)
  float* lOut = out + 71303168ull;        // (B,K,K,N)

  float*  hiW   = (float*)d_ws;           // (B*K, H) fp32
  float*  hjW   = hiW + 2097152ull;       // (B*K, H) fp32
  __bf16* zb    = (__bf16*)(hjW + 2097152ull);  // 1048576
  __bf16* w1sw  = zb + 1048576ull;        // 65536
  __bf16* w2sw  = w1sw + 65536ull;        // 32768
  __bf16* cbm   = w2sw + 32768ull;        // 4096
  __bf16* cbrow = cbm + 4096ull;          // 2048

  pack_kernel<<<1024, 256, 0, stream>>>(z, W1, W2, cb, zb, w1sw, w2sw, cbm, cbrow);
  hij_gemm_kernel<<<512, 256, 0, stream>>>(zb, w1sw, hiW, hjW);

  const size_t smem = (4096 + 4096 + 256 + 128) * sizeof(float) +
                      (32768 + 4096 + 2048 + 8 * 2048) * sizeof(__bf16);  // 144896
  dim3 grid(256, 2, 2);
  edge_fused_kernel<<<grid, 256, smem, stream>>>(hiW, hjW, b1, b2, w2sw, cbm, cbrow,
                                                 mOut, aOut, eOut, lOut);
}
